// TutorialLLM_75402445848719
// MI455X (gfx1250) — matrix operand order repaired and checked by reading, not compiled, not run on hardware
//
#include <hip/hip_runtime.h>
#include <hip/hip_bf16.h>

// Problem constants (match reference)
constexpr int BB = 2, TT = 2048, DD = 512, HH = 8, NLAYER = 4, VV = 32000;
constexpr int HDIM = DD / HH;        // 64
constexpr int BT = BB * TT;          // 4096
constexpr float EPS = 1e-5f;

typedef __attribute__((ext_vector_type(16))) _Float16 v16h;
typedef __attribute__((ext_vector_type(8)))  _Float16 v8h;
typedef __attribute__((ext_vector_type(2)))  _Float16 v2h;
typedef __attribute__((ext_vector_type(8)))  float    v8f;

#define WMMA_F16(a, b, c) \
  __builtin_amdgcn_wmma_f32_16x16x32_f16(false, (a), false, (b), (short)0, (c), false, false)

// Concatenate two v8h into a v16h (register-level, free)
#define CAT16(lo, hi) __builtin_shufflevector((lo), (hi), 0,1,2,3,4,5,6,7,8,9,10,11,12,13,14,15)
// Interleave halves of two v8h (K-pair interleave for B-matrix staging)
#define ILV_LO(lo, hi) __builtin_shufflevector((lo), (hi), 0,8,1,9,2,10,3,11)
#define ILV_HI(lo, hi) __builtin_shufflevector((lo), (hi), 4,12,5,13,6,14,7,15)

// Async global->LDS 16B copy (ASYNCcnt-tracked DMA; bypasses VGPRs).
// VDST = per-lane LDS byte address, VADDR = 64-bit global address (GV mode).
__device__ __forceinline__ void async_copy_b128(const void* lds_dst, const void* gsrc) {
  unsigned lo = (unsigned)(size_t)lds_dst;
  unsigned long long gp = (unsigned long long)(size_t)gsrc;
  asm volatile("global_load_async_to_lds_b128 %0, %1, off" :: "v"(lo), "v"(gp) : "memory");
}
__device__ __forceinline__ void async_wait0() {
  asm volatile("s_wait_asynccnt 0" ::: "memory");
}

// ---------------------------------------------------------------------------
// f32 -> f16 conversion (weights are converted once per launch)
__global__ __launch_bounds__(256)
void llm_cvt_kernel(const float* __restrict__ src, _Float16* __restrict__ dst, long long n) {
  long long stride = (long long)gridDim.x * 256;
  for (long long i = (long long)blockIdx.x * 256 + threadIdx.x; i < n; i += stride)
    dst[i] = (_Float16)src[i];
}

// ---------------------------------------------------------------------------
// Embedding: x[row] = tok_emb[token_ids[row]] + pos_emb[row % T]   (f32 residual stream)
__global__ __launch_bounds__(256)
void llm_embed_kernel(const int* __restrict__ tok, const float* __restrict__ te,
                      const float* __restrict__ pe, float* __restrict__ x) {
  int row = blockIdx.x;
  int t = row % TT;
  int id = tok[row];
  const float* ter = te + (size_t)id * DD;
  const float* per = pe + (size_t)t * DD;
  float* xr = x + (size_t)row * DD;
  for (int i = threadIdx.x; i < DD; i += 256) xr[i] = ter[i] + per[i];
}

// ---------------------------------------------------------------------------
// LayerNorm: f32 in, f16 out (feeds WMMA GEMMs only)
__global__ __launch_bounds__(256)
void llm_ln_kernel(const float* __restrict__ x, const float* __restrict__ g,
                   const float* __restrict__ b, _Float16* __restrict__ y) {
  __shared__ float r1[256], r2[256];
  int row = blockIdx.x;
  const float* xr = x + (size_t)row * DD;
  float s = 0.f, ss = 0.f;
  for (int i = threadIdx.x; i < DD; i += 256) { float v = xr[i]; s += v; ss += v * v; }
  r1[threadIdx.x] = s; r2[threadIdx.x] = ss;
  __syncthreads();
  for (int o = 128; o > 0; o >>= 1) {
    if (threadIdx.x < o) { r1[threadIdx.x] += r1[threadIdx.x + o]; r2[threadIdx.x] += r2[threadIdx.x + o]; }
    __syncthreads();
  }
  float mean = r1[0] * (1.0f / DD);
  float var  = r2[0] * (1.0f / DD) - mean * mean;
  float inv  = rsqrtf(var + EPS);
  _Float16* yr = y + (size_t)row * DD;
  for (int i = threadIdx.x; i < DD; i += 256) yr[i] = (_Float16)((xr[i] - mean) * inv * g[i] + b[i]);
}

// ---------------------------------------------------------------------------
// Tiled WMMA GEMM (f16 A,B -> f32 acc -> OutT). Block tile BM x BN, BK = 32.
// A tile staged by async global->LDS DMA (row-major; A-fragment = 2 contiguous
// b128 LDS loads). B tile staged K-pair-interleaved through registers
// (B-fragment = 8 contiguous b32 LDS loads).
// Epilogue: +bias(f32), optional relu, +residual(f32, only for OutT=float).
template <int BM, int BN, typename OutT>
__global__ __launch_bounds__((BM / 16) * (BN / 64) * 32)
void llm_gemm_kernel(const _Float16* __restrict__ A, int lda, long long sA,
                     const _Float16* __restrict__ Bm, int ldb, long long sB,
                     OutT* __restrict__ C, int ldc, long long sC,
                     const float* __restrict__ bias, long long sBias,
                     const float* __restrict__ res, long long sRes,
                     int M, int N, int K, int relu) {
  constexpr int BK = 32;
  constexpr int WCOLS = BN / 64;
  constexpr int NT = (BM / 16) * WCOLS * 32;
  __shared__ __align__(16) _Float16 As[BM][BK + 8];            // row stride 80B
  __shared__ __align__(16) _Float16 Bp[BK / 2][2 * BN + 8];    // pair-interleaved

  const int m0 = blockIdx.y * BM;
  const int n0 = blockIdx.x * BN;
  A  += (size_t)blockIdx.z * sA;
  Bm += (size_t)blockIdx.z * sB;
  C  += (size_t)blockIdx.z * sC;
  const float* biasz = bias ? bias + (size_t)blockIdx.z * sBias : nullptr;
  const float* resz  = res  ? res  + (size_t)blockIdx.z * sRes  : nullptr;

  const int tid = threadIdx.x;
  const int wave = tid >> 5, lane = tid & 31;
  const int wr = wave / WCOLS, wc = wave % WCOLS;
  const int hi16 = (lane & 16) ? 1 : 0;

  v8f acc[4] = {};

  constexpr int AITERS = (BM * BK / 8) / NT;        // exact in all instantiations
  constexpr int BITERS = ((BK / 2) * (BN / 8)) / NT;
  static_assert(AITERS * NT == BM * BK / 8, "A staging must tile exactly");
  static_assert(BITERS * NT == (BK / 2) * (BN / 8), "B staging must tile exactly");

  for (int kk = 0; kk < K; kk += BK) {
    // ---- stage A tile: async DMA straight into LDS ----
#pragma unroll
    for (int it = 0; it < AITERS; ++it) {
      const int u = tid + it * NT;
      const int r = u >> 2, c0 = (u & 3) * 8;       // BK/8 == 4
      async_copy_b128(&As[r][c0], &A[(size_t)(m0 + r) * lda + kk + c0]);
    }
    // ---- stage B tile, K-pair interleaved through registers ----
#pragma unroll
    for (int it = 0; it < BITERS; ++it) {
      const int u = tid + it * NT;
      const int k2 = u / (BN / 8), c0 = (u % (BN / 8)) * 8;
      v8h lo = *(const v8h*)&Bm[(size_t)(kk + 2 * k2) * ldb + n0 + c0];
      v8h hi = *(const v8h*)&Bm[(size_t)(kk + 2 * k2 + 1) * ldb + n0 + c0];
      *(v8h*)&Bp[k2][2 * c0]     = ILV_LO(lo, hi);
      *(v8h*)&Bp[k2][2 * c0 + 8] = ILV_HI(lo, hi);
    }
    // prefetch next K tile into cache (global_prefetch_b8)
    if (kk + BK < K) {
      __builtin_prefetch(&A[(size_t)(m0 + (tid & (BM - 1))) * lda + kk + BK], 0, 1);
      __builtin_prefetch(&Bm[(size_t)(kk + BK + (tid & (BK - 1))) * ldb + n0], 0, 1);
    }
    async_wait0();
    __syncthreads();

    // ---- A fragment: two contiguous 16B LDS loads ----
    const int mrow = wr * 16 + (lane & 15);
    const int ka0 = hi16 * 16;
    v8h alo = *(const v8h*)&As[mrow][ka0];
    v8h ahi = *(const v8h*)&As[mrow][ka0 + 8];
    v16h a = CAT16(alo, ahi);

    const int kb2 = hi16 * 8;  // pair-row base (K base 0 or 16)
#pragma unroll
    for (int jt = 0; jt < 4; ++jt) {
      const int n2 = 2 * (wc * 64 + jt * 16 + (lane & 15));
      v16h bf;
#pragma unroll
      for (int j = 0; j < 8; ++j) {
        v2h p = *(const v2h*)&Bp[kb2 + j][n2];
        bf[2 * j] = p[0]; bf[2 * j + 1] = p[1];
      }
      acc[jt] = WMMA_F16(a, bf, acc[jt]);
    }
    __syncthreads();
  }

  const int rbase = hi16 * 8;
#pragma unroll
  for (int jt = 0; jt < 4; ++jt) {
    const int col = n0 + wc * 64 + jt * 16 + (lane & 15);
#pragma unroll
    for (int r = 0; r < 8; ++r) {
      const int row = m0 + wr * 16 + rbase + r;
      float v = acc[jt][r];
      if (biasz) v += biasz[col];
      if (relu)  v = fmaxf(v, 0.f);
      if (resz)  v += resz[(size_t)row * ldc + col];
      C[(size_t)row * ldc + col] = (OutT)v;
    }
  }
}

// ---------------------------------------------------------------------------
// Flash-style causal attention: one wave per (b,h, 16-query block).
// q/k/v/o are f16, laid out (B,T,H,HD): row stride D, head offset h*HD.
__global__ __launch_bounds__(32)
void llm_attn_kernel(const _Float16* __restrict__ q, const _Float16* __restrict__ k,
                     const _Float16* __restrict__ v, _Float16* __restrict__ o) {
  __shared__ __align__(16) _Float16 Qs[16][HDIM + 8];          // 144B rows
  __shared__ __align__(16) _Float16 Ks[32][HDIM + 8];
  __shared__ __align__(16) _Float16 VsP[16][2 * HDIM + 8];     // key-pair interleaved
  __shared__ __align__(16) float    Ss[16][32];
  __shared__ __align__(16) _Float16 Ps[16][32 + 8];            // 80B rows
  __shared__ float m_s[16], l_s[16], a_s[16];

  const int bh = blockIdx.y;
  const int bb = bh / HH, h_ = bh % HH;
  const int q0 = blockIdx.x * 16;
  const int lane = threadIdx.x & 31;
  const int m = lane & 15;
  const int hi16 = (lane & 16) ? 1 : 0;
  const int rbase = hi16 * 8;
  const float scale = rsqrtf((float)DD);  // reference scales by dim_embed
  const size_t base = (size_t)bb * TT * DD + (size_t)h_ * HDIM;

  // stage Q (scaled)
#pragma unroll
  for (int it = 0; it < 4; ++it) {
    const int u = lane + it * 32;
    const int r = u >> 3, c0 = (u & 7) * 8;
    v8h t = *(const v8h*)&q[base + (size_t)(q0 + r) * DD + c0];
#pragma unroll
    for (int j = 0; j < 8; ++j) t[j] = (_Float16)((float)t[j] * scale);
    *(v8h*)&Qs[r][c0] = t;
  }
  if (lane < 16) { m_s[lane] = -__builtin_inff(); l_s[lane] = 0.f; }
  v8f oacc[4] = {};
  __syncthreads();

  const int kend = q0 + 16;  // causally reachable keys are < kend
  for (int kb = 0; kb < kend; kb += 32) {
    // stage K via async DMA (straight copy), V pair-interleaved via registers
#pragma unroll
    for (int it = 0; it < 8; ++it) {
      const int u = lane + it * 32;
      const int r = u >> 3, c0 = (u & 7) * 8;
      async_copy_b128(&Ks[r][c0], &k[base + (size_t)(kb + r) * DD + c0]);
    }
#pragma unroll
    for (int it = 0; it < 4; ++it) {
      const int u = lane + it * 32;
      const int k2 = u >> 3, c0 = (u & 7) * 8;
      v8h lo = *(const v8h*)&v[base + (size_t)(kb + 2 * k2) * DD + c0];
      v8h hi = *(const v8h*)&v[base + (size_t)(kb + 2 * k2 + 1) * DD + c0];
      *(v8h*)&VsP[k2][2 * c0]     = ILV_LO(lo, hi);
      *(v8h*)&VsP[k2][2 * c0 + 8] = ILV_HI(lo, hi);
    }
    async_wait0();
    __syncthreads();

    // S(16x32) = Q(16x64) @ K^T : contiguous fragment loads on both sides
#pragma unroll
    for (int jt = 0; jt < 2; ++jt) {
      v8f sacc = {};
#pragma unroll
      for (int kh = 0; kh < 2; ++kh) {
        const int ka0 = kh * 32 + hi16 * 16;
        v8h alo = *(const v8h*)&Qs[m][ka0];
        v8h ahi = *(const v8h*)&Qs[m][ka0 + 8];
        v16h a = CAT16(alo, ahi);
        const int kbB = kh * 32 + hi16 * 16;
        v8h blo = *(const v8h*)&Ks[jt * 16 + m][kbB];
        v8h bhi = *(const v8h*)&Ks[jt * 16 + m][kbB + 8];
        v16h bf = CAT16(blo, bhi);
        sacc = WMMA_F16(a, bf, sacc);
      }
#pragma unroll
      for (int r = 0; r < 8; ++r) Ss[rbase + r][jt * 16 + m] = sacc[r];
    }
    __syncthreads();

    // online softmax over this 32-key slab (lanes 0..15, one row each)
    if (lane < 16) {
      const int qrow = q0 + lane;
      float mx = -__builtin_inff();
      for (int j = 0; j < 32; ++j) {
        if (kb + j > qrow) Ss[lane][j] = -__builtin_inff();
        mx = fmaxf(mx, Ss[lane][j]);
      }
      float newm = fmaxf(m_s[lane], mx);
      float alpha = expf(m_s[lane] - newm);
      float sum = 0.f;
      for (int j = 0; j < 32; ++j) {
        float p = expf(Ss[lane][j] - newm);
        sum += p;
        Ps[lane][j] = (_Float16)p;
      }
      l_s[lane] = l_s[lane] * alpha + sum;
      m_s[lane] = newm;
      a_s[lane] = alpha;
    }
    __syncthreads();

    // O = O*alpha + P(16x32) @ V(32x64)
#pragma unroll
    for (int jt = 0; jt < 4; ++jt)
#pragma unroll
      for (int r = 0; r < 8; ++r) oacc[jt][r] *= a_s[rbase + r];
    {
      const int ka0 = hi16 * 16;
      v8h alo = *(const v8h*)&Ps[m][ka0];
      v8h ahi = *(const v8h*)&Ps[m][ka0 + 8];
      v16h a = CAT16(alo, ahi);
      const int kb2 = hi16 * 8;
#pragma unroll
      for (int jt = 0; jt < 4; ++jt) {
        const int n2 = 2 * (jt * 16 + m);
        v16h bf;
#pragma unroll
        for (int j = 0; j < 8; ++j) {
          v2h p = *(const v2h*)&VsP[kb2 + j][n2];
          bf[2 * j] = p[0]; bf[2 * j + 1] = p[1];
        }
        oacc[jt] = WMMA_F16(a, bf, oacc[jt]);
      }
    }
    __syncthreads();
  }

#pragma unroll
  for (int jt = 0; jt < 4; ++jt)
#pragma unroll
    for (int r = 0; r < 8; ++r) {
      const int qrow = q0 + rbase + r;
      o[base + (size_t)qrow * DD + jt * 16 + m] = (_Float16)(oacc[jt][r] / l_s[rbase + r]);
    }
}

// ---------------------------------------------------------------------------
// NLL loss over logits rows (logits left untouched in d_out)
__global__ __launch_bounds__(256)
void llm_loss_kernel(const float* __restrict__ logits, const int* __restrict__ labels,
                     float* __restrict__ loss) {
  __shared__ float red[256];
  const int row = blockIdx.x;
  const float* lr = logits + (size_t)row * VV;
  float mx = -__builtin_inff();
  for (int i = threadIdx.x; i < VV; i += 256) mx = fmaxf(mx, lr[i]);
  red[threadIdx.x] = mx;
  __syncthreads();
  for (int o = 128; o > 0; o >>= 1) {
    if (threadIdx.x < o) red[threadIdx.x] = fmaxf(red[threadIdx.x], red[threadIdx.x + o]);
    __syncthreads();
  }
  mx = red[0];
  __syncthreads();
  float s = 0.f;
  for (int i = threadIdx.x; i < VV; i += 256) s += expf(lr[i] - mx);
  red[threadIdx.x] = s;
  __syncthreads();
  for (int o = 128; o > 0; o >>= 1) {
    if (threadIdx.x < o) red[threadIdx.x] += red[threadIdx.x + o];
    __syncthreads();
  }
  if (threadIdx.x == 0) {
    float lse = mx + logf(red[0]);
    float contrib = (lse - lr[labels[row]]) * (1.0f / (float)BT);
    atomicAdd(loss, contrib);
  }
}

// ---------------------------------------------------------------------------
extern "C" void kernel_launch(void* const* d_in, const int* in_sizes, int n_in,
                              void* d_out, int out_size, void* d_ws, size_t ws_size,
                              hipStream_t stream) {
  const int*   token_ids = (const int*)d_in[0];
  const int*   labels    = (const int*)d_in[1];
  const float* tok_emb   = (const float*)d_in[2];
  const float* pos_emb   = (const float*)d_in[3];
  const float* Wq        = (const float*)d_in[4];
  const float* Wk        = (const float*)d_in[5];
  const float* Wv        = (const float*)d_in[6];
  const float* Wo        = (const float*)d_in[7];
  const float* bo        = (const float*)d_in[8];
  const float* ln1_g     = (const float*)d_in[9];
  const float* ln1_b     = (const float*)d_in[10];
  const float* ln2_g     = (const float*)d_in[11];
  const float* ln2_b     = (const float*)d_in[12];
  const float* W1        = (const float*)d_in[13];
  const float* b1        = (const float*)d_in[14];
  const float* W2        = (const float*)d_in[15];
  const float* b2        = (const float*)d_in[16];
  const float* lnf_g     = (const float*)d_in[17];
  const float* lnf_b     = (const float*)d_in[18];
  const float* Wp        = (const float*)d_in[19];
  const float* bp        = (const float*)d_in[20];

  float* out = (float*)d_out;                 // [BT*V logits][1 loss]
  const long long NXD  = (long long)BT * DD;  // 2,097,152
  const long long NQKV = (long long)NLAYER * HH * DD * HDIM;   // 1,048,576
  const long long NWO  = (long long)NLAYER * DD * DD;          // 1,048,576
  const long long NW1  = (long long)NLAYER * DD * 4 * DD;      // 4,194,304
  const long long NWP  = (long long)DD * VV;                   // 16,384,000

  float*    x   = (float*)d_ws;               // f32 residual stream
  _Float16* hb  = (_Float16*)(x + NXD);
  _Float16* xn  = hb;            hb += NXD;
  _Float16* qh  = hb;            hb += NXD;
  _Float16* kh  = hb;            hb += NXD;
  _Float16* vh  = hb;            hb += NXD;
  _Float16* oh  = hb;            hb += NXD;
  _Float16* h1h = hb;            hb += (long long)BT * 4 * DD;
  _Float16* wqh = hb;            hb += NQKV;
  _Float16* wkh = hb;            hb += NQKV;
  _Float16* wvh = hb;            hb += NQKV;
  _Float16* woh = hb;            hb += NWO;
  _Float16* w1h = hb;            hb += NW1;
  _Float16* w2h = hb;            hb += NW1;
  _Float16* wph = hb;

  // one-time (per launch) f16 weight conversion
  llm_cvt_kernel<<<1024, 256, 0, stream>>>(Wq, wqh, NQKV);
  llm_cvt_kernel<<<1024, 256, 0, stream>>>(Wk, wkh, NQKV);
  llm_cvt_kernel<<<1024, 256, 0, stream>>>(Wv, wvh, NQKV);
  llm_cvt_kernel<<<1024, 256, 0, stream>>>(Wo, woh, NWO);
  llm_cvt_kernel<<<1024, 256, 0, stream>>>(W1, w1h, NW1);
  llm_cvt_kernel<<<1024, 256, 0, stream>>>(W2, w2h, NW1);
  llm_cvt_kernel<<<2048, 256, 0, stream>>>(Wp, wph, NWP);

  // x = tok_emb[ids] + pos_emb
  llm_embed_kernel<<<BT, 256, 0, stream>>>(token_ids, tok_emb, pos_emb, x);

  for (int l = 0; l < NLAYER; ++l) {
    // xn = LN1(x)  (f16)
    llm_ln_kernel<<<BT, 256, 0, stream>>>(x, ln1_g + (size_t)l * DD, ln1_b + (size_t)l * DD, xn);

    // q/k/v (head-batched, N = HD = 64), f16 out
    const long long sW = (long long)DD * HDIM;
    dim3 gq(1, BT / 64, HH);
    llm_gemm_kernel<64, 64, _Float16><<<gq, 128, 0, stream>>>(
        xn, DD, 0LL, wqh + (size_t)l * HH * DD * HDIM, HDIM, sW,
        qh, DD, (long long)HDIM, nullptr, 0LL, nullptr, 0LL, BT, HDIM, DD, 0);
    llm_gemm_kernel<64, 64, _Float16><<<gq, 128, 0, stream>>>(
        xn, DD, 0LL, wkh + (size_t)l * HH * DD * HDIM, HDIM, sW,
        kh, DD, (long long)HDIM, nullptr, 0LL, nullptr, 0LL, BT, HDIM, DD, 0);
    llm_gemm_kernel<64, 64, _Float16><<<gq, 128, 0, stream>>>(
        xn, DD, 0LL, wvh + (size_t)l * HH * DD * HDIM, HDIM, sW,
        vh, DD, (long long)HDIM, nullptr, 0LL, nullptr, 0LL, BT, HDIM, DD, 0);

    // causal attention -> oh (B,T,H,HD) == concat-heads layout, f16
    llm_attn_kernel<<<dim3(TT / 16, BB * HH), 32, 0, stream>>>(qh, kh, vh, oh);

    // x = x + oh @ Wo + bo   (f32 residual out)
    llm_gemm_kernel<64, 128, float><<<dim3(DD / 128, BT / 64), 256, 0, stream>>>(
        oh, DD, 0LL, woh + (size_t)l * DD * DD, DD, 0LL,
        x, DD, 0LL, bo + (size_t)l * DD, 0LL, x, 0LL, BT, DD, DD, 0);

    // xn = LN2(x)
    llm_ln_kernel<<<BT, 256, 0, stream>>>(x, ln2_g + (size_t)l * DD, ln2_b + (size_t)l * DD, xn);

    // h1 = relu(xn @ W1 + b1)  (f16 out)
    llm_gemm_kernel<64, 128, _Float16><<<dim3(4 * DD / 128, BT / 64), 256, 0, stream>>>(
        xn, DD, 0LL, w1h + (size_t)l * DD * 4 * DD, 4 * DD, 0LL,
        h1h, 4 * DD, 0LL, b1 + (size_t)l * 4 * DD, 0LL, nullptr, 0LL, BT, 4 * DD, DD, 1);

    // x = x + h1 @ W2 + b2   (f32 residual out)
    llm_gemm_kernel<64, 128, float><<<dim3(DD / 128, BT / 64), 256, 0, stream>>>(
        h1h, 4 * DD, 0LL, w2h + (size_t)l * 4 * DD * DD, DD, 0LL,
        x, DD, 0LL, b2 + (size_t)l * DD, 0LL, x, 0LL, BT, DD, 4 * DD, 0);
  }

  // xn = LNf(x); logits = xn @ Wp + bp  -> d_out (f32)
  llm_ln_kernel<<<BT, 256, 0, stream>>>(x, lnf_g, lnf_b, xn);
  llm_gemm_kernel<64, 128, float><<<dim3(VV / 128, BT / 64), 256, 0, stream>>>(
      xn, DD, 0LL, wph, VV, 0LL,
      out, VV, 0LL, bp, 0LL, nullptr, 0LL, BT, VV, DD, 0);

  // loss = mean NLL
  hipMemsetAsync(out + (size_t)BT * VV, 0, sizeof(float), stream);
  llm_loss_kernel<<<BT, 256, 0, stream>>>(out, labels, out + (size_t)BT * VV);
}
